// PreT_Attention_9234179687674
// MI455X (gfx1250) — matrix operand order repaired and checked by reading, not compiled, not run on hardware
//
#include <hip/hip_runtime.h>
#include <hip/hip_bf16.h>
#include <stddef.h>
#include <stdint.h>

// ---------------------------------------------------------------------------
// PreT attention for MI455X (gfx1250): bf16 WMMA + async global->LDS staging.
//   B=8 N=1024 C=1024 H=16 P=64 HD=64, M_kv = P+N = 1088
// ---------------------------------------------------------------------------

typedef unsigned short u16;
typedef u16   v8u16  __attribute__((ext_vector_type(8)));
typedef u16   v16u16 __attribute__((ext_vector_type(16)));
typedef __bf16 v16bf __attribute__((ext_vector_type(16)));
typedef float v8f    __attribute__((ext_vector_type(8)));

#define WMMA_BF16(a, b, c) \
  __builtin_amdgcn_wmma_f32_16x16x32_bf16(false, (a), false, (b), (short)0, (c), false, false)

static constexpr int Bb = 8, Nn = 1024, Cc = 1024, Hh = 16, Pp = 64, HDd = 64;
static constexpr int Mkv = Pp + Nn;          // 1088
static constexpr int ASTR = 40;              // padded LDS stride (bf16) for 32-wide tiles
static constexpr int KSTR = 72;              // padded LDS stride for 64-wide K rows
static constexpr int VSTR = 40;              // padded LDS stride for 32-wide V^T rows

// fp32 -> bf16, round-half-up (1 add + shift; bias limited to ties).
__device__ __forceinline__ u16 f2bf(float f) {
  unsigned u = __builtin_bit_cast(unsigned, f);
  return (u16)((u + 0x8000u) >> 16);
}

// Pack two fp32 into bf16x2 with one v_perm_b32: result = {hi16(b+r), hi16(a+r)}.
// perm sel: result bytes [a.b2, a.b3, b.b2, b.b3] -> 0x03020706 (S0=a codes 4-7).
__device__ __forceinline__ unsigned pack2bf(float a, float b) {
  unsigned ra = __builtin_bit_cast(unsigned, a) + 0x8000u;
  unsigned rb = __builtin_bit_cast(unsigned, b) + 0x8000u;
  return __builtin_amdgcn_perm(ra, rb, 0x03020706u);
}

// Build a v16bf WMMA fragment from two contiguous 8-element (16B) chunks.
__device__ __forceinline__ v16bf frag_from(const u16* p0, const u16* p1) {
  v8u16 lo = *(const v8u16*)p0;
  v8u16 hi = *(const v8u16*)p1;
  v16u16 t = __builtin_shufflevector(lo, hi, 0,1,2,3,4,5,6,7,8,9,10,11,12,13,14,15);
  return __builtin_bit_cast(v16bf, t);
}

__device__ __forceinline__ void ld4f4(float4* d, const float* src) {
  const float4* s4 = (const float4*)src;
#pragma unroll
  for (int i = 0; i < 4; i++) d[i] = s4[i];
}

__device__ __forceinline__ void st_cvt16(u16* dst, const float4* f) {
#pragma unroll
  for (int i = 0; i < 4; i++) {
    uint2 pk;
    pk.x = pack2bf(f[i].x, f[i].y);
    pk.y = pack2bf(f[i].z, f[i].w);
    *(uint2*)(dst + i * 4) = pk;
  }
}

// ---- CDNA5 async global -> LDS (ASYNCcnt-tracked, bypasses VGPRs) ----------
__device__ __forceinline__ unsigned lds_off(const void* p) {
  return (unsigned)(uintptr_t)p;    // generic LDS addr: low 32 bits = LDS offset
}
__device__ __forceinline__ void async_ld_b128(unsigned ldsaddr, const void* gaddr) {
  asm volatile("global_load_async_to_lds_b128 %0, %1, off"
               :: "v"(ldsaddr), "v"(gaddr) : "memory");
}
__device__ __forceinline__ void wait_async0() {
  asm volatile("s_wait_asynccnt 0" ::: "memory");
}

// ---------------------------------------------------------------------------
// Kernel 0: prompt (B,2,P,H,HD) fp32 -> bf16 K/V prefix rows [0,P).
// ---------------------------------------------------------------------------
__global__ __launch_bounds__(256) void prefill_kv(const float* __restrict__ prompt,
                                                  u16* __restrict__ kbuf,
                                                  u16* __restrict__ vbuf) {
  size_t i = (size_t)blockIdx.x * 256 + threadIdx.x;
  int d = (int)(i & 63);
  size_t r = i >> 6;
  int h = (int)(r & 15); r >>= 4;
  int p = (int)(r & 63); r >>= 6;
  int tt = (int)(r & 1);
  int b = (int)(r >> 1);
  u16 val = f2bf(prompt[i]);
  u16* dst = tt ? vbuf : kbuf;
  dst[(((size_t)b * Hh + h) * Mkv + p) * HDd + d] = val;
}

// ---------------------------------------------------------------------------
// Kernel 1: qkv = x @ w_qkv^T, scatter to q/k/v bf16 buffers.
// 128x128 tile, double-buffered LDS, one barrier per K-step.
// ---------------------------------------------------------------------------
__global__ __launch_bounds__(256) void qkv_gemm(const float* __restrict__ x,
                                                const float* __restrict__ wqkv,
                                                u16* __restrict__ qbuf,
                                                u16* __restrict__ kbuf,
                                                u16* __restrict__ vbuf) {
  __shared__ u16 ldsA[2][128 * ASTR];
  __shared__ u16 ldsB[2][128 * ASTR];
  const int t  = threadIdx.x;
  const int m0 = blockIdx.y * 128;
  const int n0 = blockIdx.x * 128;
  const int w  = t >> 5;
  const int ln = t & 15;
  const int hi = (t >> 4) & 1;
  const int wm = (w & 3) * 32;
  const int wn = (w >> 2) * 64;
  const int row = t >> 1;
  const int ko  = (t & 1) * 16;
  const float* srcA = x    + (size_t)(m0 + row) * Cc + ko;
  const float* srcB = wqkv + (size_t)(n0 + row) * Cc + ko;

  v8f acc[2][4];
#pragma unroll
  for (int mt = 0; mt < 2; mt++)
#pragma unroll
    for (int nt = 0; nt < 4; nt++) acc[mt][nt] = {};

  float4 pa[4], pb[4];
  ld4f4(pa, srcA);
  ld4f4(pb, srcB);
  st_cvt16(&ldsA[0][row * ASTR + ko], pa);
  st_cvt16(&ldsB[0][row * ASTR + ko], pb);
  __syncthreads();

  const int NSTEP = Cc / 32;
  for (int s = 0; s < NSTEP; s++) {
    const int cur = s & 1;
    const bool more = (s + 1 < NSTEP);
    if (more) {                       // issue next tile's global loads now
      ld4f4(pa, srcA + (s + 1) * 32);
      ld4f4(pb, srcB + (s + 1) * 32);
    }
    v16bf af[2], bfr[4];
#pragma unroll
    for (int mt = 0; mt < 2; mt++) {
      const u16* p = &ldsA[cur][(wm + mt * 16 + ln) * ASTR];
      af[mt] = frag_from(p + hi * 16, p + hi * 16 + 8);
    }
#pragma unroll
    for (int nt = 0; nt < 4; nt++) {
      const u16* p = &ldsB[cur][(wn + nt * 16 + ln) * ASTR + hi * 16];
      bfr[nt] = frag_from(p, p + 8);
    }
#pragma unroll
    for (int nt = 0; nt < 4; nt++)
#pragma unroll
      for (int mt = 0; mt < 2; mt++)
        acc[mt][nt] = WMMA_BF16(af[mt], bfr[nt], acc[mt][nt]);
    if (more) {                       // fill the other buffer
      st_cvt16(&ldsA[cur ^ 1][row * ASTR + ko], pa);
      st_cvt16(&ldsB[cur ^ 1][row * ASTR + ko], pb);
    }
    __syncthreads();
  }

  // epilogue: q/k/v destination is uniform per 16-wide column tile -> hoist.
#pragma unroll
  for (int nt = 0; nt < 4; nt++) {
    const int gn = n0 + wn + nt * 16 + ln;
    const int tsel = gn >> 10, h = (gn >> 6) & 15, d = gn & 63;
#pragma unroll
    for (int mt = 0; mt < 2; mt++) {
      const int gmb = m0 + wm + mt * 16 + 8 * hi;   // rows gmb..gmb+7, no 1024-cross
      const int b = gmb >> 10, nr = gmb & 1023;
      u16* dst;
      if (tsel == 0)
        dst = qbuf + (((size_t)b * Hh + h) * Nn + nr) * HDd + d;
      else if (tsel == 1)
        dst = kbuf + (((size_t)b * Hh + h) * Mkv + Pp + nr) * HDd + d;
      else
        dst = vbuf + (((size_t)b * Hh + h) * Mkv + Pp + nr) * HDd + d;
#pragma unroll
      for (int r = 0; r < 8; r++) dst[(size_t)r * HDd] = f2bf(acc[mt][nt][r]);
    }
  }
}

// ---------------------------------------------------------------------------
// Kernel 2: flash attention. Double-buffered K/V chunks; K staged with
// async global->LDS (ASYNCcnt), V transposed through VGPRs.
// The PV contraction index is stored permuted: key j -> column
// sigma(j) = 2*(j&15) + (j>>4), applied to BOTH P and V^T, so each lane's
// two probabilities pack into one b32 LDS store.
// ---------------------------------------------------------------------------
__global__ __launch_bounds__(256) void attn(const u16* __restrict__ qbuf,
                                            const u16* __restrict__ kbuf,
                                            const u16* __restrict__ vbuf,
                                            u16* __restrict__ aobuf) {
  __shared__ u16 ldsK[2][32 * KSTR];    // K chunk, row-major [krow][d]
  __shared__ u16 ldsVt[2][64 * VSTR];   // V chunk transposed [d][sigma(krow)]
  __shared__ u16 ldsP[8 * 16 * 32];     // per-wave P relayout scratch

  const int t  = threadIdx.x;
  const int w  = t >> 5;
  const int ln = t & 15;
  const int hi = (t >> 4) & 1;
  const int bh = blockIdx.y;
  const int qbase = blockIdx.x * 128;
  const float scale = 0.125f;           // HD^-0.5

  const u16* qp = qbuf + ((size_t)bh * Nn + qbase + w * 16 + ln) * HDd;
  v16bf qa0 = frag_from(qp + hi * 16,      qp + hi * 16 + 8);
  v16bf qa1 = frag_from(qp + 32 + hi * 16, qp + 40 + hi * 16);

  u16* pB = &ldsP[w * 512];

  v8f o[4];
#pragma unroll
  for (int f = 0; f < 4; f++) o[f] = {};
  float rm[8], rs[8];
#pragma unroll
  for (int r = 0; r < 8; r++) { rm[r] = -1e30f; rs[r] = 0.0f; }

  const int srow = t >> 3;                       // 0..31 key row in chunk
  const int scol = ((srow & 15) << 1) | (srow >> 4);  // sigma(srow)
  const int sdo  = (t & 7) * 8;                  // 8 bf16 per thread
  const u16* kg = kbuf + ((size_t)bh * Mkv + srow) * HDd + sdo;
  const u16* vg = vbuf + ((size_t)bh * Mkv + srow) * HDd + sdo;
  const int CH = Mkv / 32;                       // 34

  // prologue: stage chunk 0 into buffer 0
  {
    async_ld_b128(lds_off(&ldsK[0][srow * KSTR + sdo]), kg);
    v8u16 vv = *(const v8u16*)vg;
#pragma unroll
    for (int i = 0; i < 8; i++) ldsVt[0][(sdo + i) * VSTR + scol] = vv[i];
    wait_async0();
    __syncthreads();
  }

  for (int c = 0; c < CH; c++) {
    const int cur = c & 1;
    const bool more = (c + 1 < CH);
    v8u16 vv = {};
    if (more) {
      async_ld_b128(lds_off(&ldsK[cur ^ 1][srow * KSTR + sdo]),
                    kg + (size_t)(c + 1) * 32 * HDd);
      vv = *(const v8u16*)(vg + (size_t)(c + 1) * 32 * HDd);
      if (c + 2 < CH)
        __builtin_prefetch(vg + (size_t)(c + 2) * 32 * HDd, 0, 0);
    }

    // --- S = Q K^T for keys [c*32, c*32+32)
    v8f s0 = {}, s1 = {};
    {
      const u16* p0 = &ldsK[cur][ln * KSTR + hi * 16];
      s0 = WMMA_BF16(qa0, frag_from(p0,      p0 + 8),  s0);
      s0 = WMMA_BF16(qa1, frag_from(p0 + 32, p0 + 40), s0);
      const u16* p1 = &ldsK[cur][(16 + ln) * KSTR + hi * 16];
      s1 = WMMA_BF16(qa0, frag_from(p1,      p1 + 8),  s1);
      s1 = WMMA_BF16(qa1, frag_from(p1 + 32, p1 + 40), s1);
    }

    // --- online softmax per C-slot row (row = r + 8*hi)
#pragma unroll
    for (int r = 0; r < 8; r++) {
      float x0 = s0[r] * scale, x1 = s1[r] * scale;
      float cm = fmaxf(x0, x1);
      cm = fmaxf(cm, __shfl_xor(cm, 1, 32));
      cm = fmaxf(cm, __shfl_xor(cm, 2, 32));
      cm = fmaxf(cm, __shfl_xor(cm, 4, 32));
      cm = fmaxf(cm, __shfl_xor(cm, 8, 32));
      float nm = fmaxf(rm[r], cm);
      float corr = __expf(rm[r] - nm);
      rm[r] = nm;
      float p0 = __expf(x0 - nm), p1 = __expf(x1 - nm);
      float ps = p0 + p1;
      ps += __shfl_xor(ps, 1, 32);
      ps += __shfl_xor(ps, 2, 32);
      ps += __shfl_xor(ps, 4, 32);
      ps += __shfl_xor(ps, 8, 32);
      rs[r] = rs[r] * corr + ps;
#pragma unroll
      for (int f = 0; f < 4; f++) o[f][r] *= corr;
      // keys ln (tile0) and 16+ln (tile1) land at columns 2*ln, 2*ln+1
      int mrow = r + 8 * hi;
      *(unsigned*)&pB[mrow * 32 + 2 * ln] = pack2bf(p0, p1);
    }

    // --- O += P (16x32) @ V (32x64)  [K-dim permuted by sigma in both]
    const u16* pa = &pB[ln * 32];
    v16bf paf = frag_from(pa + hi * 16, pa + hi * 16 + 8);
#pragma unroll
    for (int f = 0; f < 4; f++) {
      const u16* q = &ldsVt[cur][(f * 16 + ln) * VSTR + hi * 16];
      o[f] = WMMA_BF16(paf, frag_from(q, q + 8), o[f]);
    }

    if (more) {
#pragma unroll
      for (int i = 0; i < 8; i++) ldsVt[cur ^ 1][(sdo + i) * VSTR + scol] = vv[i];
    }
    wait_async0();
    __syncthreads();
  }

  // epilogue: normalize and store bf16 (B,N,C)
  const int b = bh >> 4, h = bh & 15;
#pragma unroll
  for (int r = 0; r < 8; r++) {
    float inv = 1.0f / rs[r];
    int qrow = qbase + w * 16 + r + 8 * hi;
    u16* op = aobuf + ((size_t)b * Nn + qrow) * Cc + h * 64 + ln;
    op[0]  = f2bf(o[0][r] * inv);
    op[16] = f2bf(o[1][r] * inv);
    op[32] = f2bf(o[2][r] * inv);
    op[48] = f2bf(o[3][r] * inv);
  }
}

// ---------------------------------------------------------------------------
// Kernel 3: out = attn_out @ w_proj^T + b_proj (fp32 out).
// A (bf16) staged with async global->LDS; B converted through VGPRs.
// ---------------------------------------------------------------------------
__global__ __launch_bounds__(256) void proj_gemm(const u16* __restrict__ aobuf,
                                                 const float* __restrict__ wproj,
                                                 const float* __restrict__ bproj,
                                                 float* __restrict__ out) {
  __shared__ u16 ldsA[2][128 * ASTR];
  __shared__ u16 ldsB[2][128 * ASTR];
  const int t  = threadIdx.x;
  const int m0 = blockIdx.y * 128;
  const int n0 = blockIdx.x * 128;
  const int w  = t >> 5;
  const int ln = t & 15;
  const int hi = (t >> 4) & 1;
  const int wm = (w & 3) * 32;
  const int wn = (w >> 2) * 64;
  const int row = t >> 1;
  const int ko  = (t & 1) * 16;
  const u16*   srcA = aobuf + (size_t)(m0 + row) * Cc + ko;
  const float* srcB = wproj + (size_t)(n0 + row) * Cc + ko;

  v8f acc[2][4];
#pragma unroll
  for (int mt = 0; mt < 2; mt++)
#pragma unroll
    for (int nt = 0; nt < 4; nt++) acc[mt][nt] = {};

  float4 pb[4];
  async_ld_b128(lds_off(&ldsA[0][row * ASTR + ko]),     srcA);
  async_ld_b128(lds_off(&ldsA[0][row * ASTR + ko + 8]), srcA + 8);
  ld4f4(pb, srcB);
  st_cvt16(&ldsB[0][row * ASTR + ko], pb);
  wait_async0();
  __syncthreads();

  const int NSTEP = Cc / 32;
  for (int s = 0; s < NSTEP; s++) {
    const int cur = s & 1;
    const bool more = (s + 1 < NSTEP);
    if (more) {
      async_ld_b128(lds_off(&ldsA[cur ^ 1][row * ASTR + ko]),     srcA + (s + 1) * 32);
      async_ld_b128(lds_off(&ldsA[cur ^ 1][row * ASTR + ko + 8]), srcA + (s + 1) * 32 + 8);
      ld4f4(pb, srcB + (s + 1) * 32);
    }
    v16bf af[2], bfr[4];
#pragma unroll
    for (int mt = 0; mt < 2; mt++) {
      const u16* p = &ldsA[cur][(wm + mt * 16 + ln) * ASTR];
      af[mt] = frag_from(p + hi * 16, p + hi * 16 + 8);
    }
#pragma unroll
    for (int nt = 0; nt < 4; nt++) {
      const u16* p = &ldsB[cur][(wn + nt * 16 + ln) * ASTR + hi * 16];
      bfr[nt] = frag_from(p, p + 8);
    }
#pragma unroll
    for (int nt = 0; nt < 4; nt++)
#pragma unroll
      for (int mt = 0; mt < 2; mt++)
        acc[mt][nt] = WMMA_BF16(af[mt], bfr[nt], acc[mt][nt]);
    if (more) st_cvt16(&ldsB[cur ^ 1][row * ASTR + ko], pb);
    wait_async0();
    __syncthreads();
  }

#pragma unroll
  for (int nt = 0; nt < 4; nt++) {
    const int gn = n0 + wn + nt * 16 + ln;
    const float bias = bproj[gn];
#pragma unroll
    for (int mt = 0; mt < 2; mt++) {
      const int gm = m0 + wm + mt * 16 + 8 * hi;
      float* op = out + (size_t)gm * Cc + gn;
#pragma unroll
      for (int r = 0; r < 8; r++) op[(size_t)r * Cc] = acc[mt][nt][r] + bias;
    }
  }
}

// ---------------------------------------------------------------------------
// Launch. Inputs: x, prompt, w_qkv, w_proj, b_proj (all fp32).
// Workspace (bf16): q (16MB) | k (17MB) | v (17MB) | attn_out (16MB) ~= 69MB.
// ---------------------------------------------------------------------------
extern "C" void kernel_launch(void* const* d_in, const int* in_sizes, int n_in,
                              void* d_out, int out_size, void* d_ws, size_t ws_size,
                              hipStream_t stream) {
  (void)in_sizes; (void)n_in; (void)out_size; (void)ws_size;
  const float* x      = (const float*)d_in[0];
  const float* prompt = (const float*)d_in[1];
  const float* wqkv   = (const float*)d_in[2];
  const float* wproj  = (const float*)d_in[3];
  const float* bproj  = (const float*)d_in[4];
  float* out = (float*)d_out;

  u16* qbuf  = (u16*)d_ws;
  u16* kbuf  = qbuf + (size_t)Bb * Hh * Nn * HDd;
  u16* vbuf  = kbuf + (size_t)Bb * Hh * Mkv * HDd;
  u16* aobuf = vbuf + (size_t)Bb * Hh * Mkv * HDd;

  prefill_kv<<<(Bb * 2 * Pp * Hh * HDd) / 256, 256, 0, stream>>>(prompt, kbuf, vbuf);
  qkv_gemm<<<dim3(3 * Cc / 128, Bb * Nn / 128), 256, 0, stream>>>(x, wqkv, qbuf, kbuf, vbuf);
  attn<<<dim3(Nn / 128, Bb * Hh), 256, 0, stream>>>(qbuf, kbuf, vbuf, aobuf);
  proj_gemm<<<dim3(Cc / 128, Bb * Nn / 128), 256, 0, stream>>>(aobuf, wproj, bproj, out);
}
